// DualAgentEnvironment_79096117723540
// MI455X (gfx1250) — matrix operand order repaired and checked by reading, not compiled, not run on hardware
//
#include <hip/hip_runtime.h>
#include <hip/hip_bf16.h>
#include <math.h>

// ---------------------------------------------------------------------------
// Problem dims (fixed by the reference)
// ---------------------------------------------------------------------------
#define B_DIM 4096
#define T_DIM 64
#define F_DIM 128
#define H_DIM 128
#define NROW  (B_DIM * T_DIM)   // 262144 rows
#define LAM   0.5f

// LDS tile: 16 rows x 512 usable cols (4 regions of 128), stride padded to 516
// dwords so row-to-row bank advance is 4 -> conflict-free column access on the
// 64-bank LDS.
#define NC 516

typedef __attribute__((ext_vector_type(16))) __bf16 v16bf;
typedef __attribute__((ext_vector_type(8)))  float  v8f;
typedef unsigned short u16;

// f32 -> bf16 RNE bit-twiddle (used only in the one-shot weight repack)
static __device__ __forceinline__ u16 f2bf_bits(float f) {
  unsigned u = __float_as_uint(f);
  u += 0x7FFFu + ((u >> 16) & 1u);
  return (u16)(u >> 16);
}

// exact GELU (matches jax.nn.gelu(approximate=False))
static __device__ __forceinline__ float gelu_exact(float x) {
  return 0.5f * x * (1.0f + erff(x * 0.70710678118654752440f));
}

// ---------------------------------------------------------------------------
// Weight repack: f32 row-major W[K][128] -> bf16 packed in WMMA-B lane layout.
// For 16x16x32 bf16 WMMA, B is 32x16 per (K-chunk c, N-tile nt):
//   lanes 0-15 : elems 0-7 hold K = c*32+0..7,  elems 8-15 K = c*32+16..23, N = lane
//   lanes 16-31: elems 0-7 hold K = c*32+8..15, elems 8-15 K = c*32+24..31, N = lane-16
// Flat packed index: ((nt*KC + c)*32 + lane)*16 + i   (16 contiguous bf16/lane
// => each WMMA B fetch is two global_load_b128 from L2-resident memory).
// ---------------------------------------------------------------------------
__global__ __launch_bounds__(256) void pack_weight(
    const float* __restrict__ W, u16* __restrict__ P, int KC /* = K/32 */)
{
  const int idx = blockIdx.x * 256 + threadIdx.x;
  const int total = KC * 32 * 128; // K * N
  if (idx >= total) return;
  const int i    = idx & 15;
  const int lane = (idx >> 4) & 31;
  const int rest = idx >> 9;
  const int c    = rest % KC;
  const int nt   = rest / KC;
  const int k    = c * 32 + (lane >> 4) * 8 + (i & 7) + (i >> 3) * 16;
  const int n    = nt * 16 + (lane & 15);
  P[idx] = f2bf_bits(W[k * 128 + n]);
}

// ---------------------------------------------------------------------------
// One 16-row x (32*KC -> 128) dense layer on a single wave.
//   src: LDS base (buf + srcCol), 16 rows, stride NC, f32 activations
//   Wp : packed bf16 weights (global, L2-resident)
//   dst: LDS base (buf + dstCol), bias added, optional exact GELU
// A-operand layout per ISA (16-bit A 16x32):
//   lanes 0-15 : elems 0-7 = K c*32+0..7,  elems 8-15 = K c*32+16..23, M=lane
//   lanes 16-31: elems 0-7 = K c*32+8..15, elems 8-15 = K c*32+24..31
// D (C-layout): VGPR r -> row r (lanes 0-15) / row r+8 (lanes 16-31),
//               col = nt*16 + (lane & 15).
// A conversion uses native casts so clang emits packed v_cvt_pk_bf16_f32.
// ---------------------------------------------------------------------------
template<int KC, bool ACT>
static __device__ __forceinline__ void wave_layer(
    const float* __restrict__ src, const u16* __restrict__ Wp,
    const float* __restrict__ bias, float* __restrict__ dst, int lane)
{
  const int row = lane & 15;
  const int hi  = lane >> 4;

  v8f zero = {0.f, 0.f, 0.f, 0.f, 0.f, 0.f, 0.f, 0.f};
  v8f acc[8];
#pragma unroll
  for (int nt = 0; nt < 8; ++nt) acc[nt] = zero;

#pragma unroll
  for (int c = 0; c < KC; ++c) {
    // Build A chunk for this lane: two contiguous groups of 8 f32 from LDS.
    const float4* q = (const float4*)(src + row * NC + c * 32 + hi * 8);
    const float4 a0 = q[0], a1 = q[1];   // K base + 0..7
    const float4 a2 = q[4], a3 = q[5];   // K base + 16..23
    v16bf A;
    A[0]  = (__bf16)a0.x; A[1]  = (__bf16)a0.y; A[2]  = (__bf16)a0.z; A[3]  = (__bf16)a0.w;
    A[4]  = (__bf16)a1.x; A[5]  = (__bf16)a1.y; A[6]  = (__bf16)a1.z; A[7]  = (__bf16)a1.w;
    A[8]  = (__bf16)a2.x; A[9]  = (__bf16)a2.y; A[10] = (__bf16)a2.z; A[11] = (__bf16)a2.w;
    A[12] = (__bf16)a3.x; A[13] = (__bf16)a3.y; A[14] = (__bf16)a3.z; A[15] = (__bf16)a3.w;

    const u16* wb = Wp + ((size_t)c * 32 + lane) * 16;
#pragma unroll
    for (int nt = 0; nt < 8; ++nt) {
      v16bf Bv = *(const v16bf*)(wb + (size_t)nt * KC * 512);
      acc[nt] = __builtin_amdgcn_wmma_f32_16x16x32_bf16(
          /*neg_a=*/false, A, /*neg_b=*/false, Bv,
          /*c_mod=*/(short)0, acc[nt], /*reuse_a=*/false, /*reuse_b=*/false);
    }
  }

  // Scatter D (C-layout) back to LDS with bias (+GELU)
#pragma unroll
  for (int nt = 0; nt < 8; ++nt) {
    const int col  = nt * 16 + row;
    const float bc = bias[col];
#pragma unroll
    for (int r = 0; r < 8; ++r) {
      float v = acc[nt][r] + bc;
      if (ACT) v = gelu_exact(v);
      dst[(r + hi * 8) * NC + col] = v;
    }
  }
}

// ---------------------------------------------------------------------------
// Row-wise LayerNorm over W cols (W = 128 or 256): two lanes per row
// (halves combined with a cross-half shuffle on the wave32).
// ---------------------------------------------------------------------------
template<int W>
static __device__ __forceinline__ void wave_ln(
    const float* __restrict__ src, float* __restrict__ dst,
    const float* __restrict__ g, const float* __restrict__ beta, int lane)
{
  const int row = lane & 15;
  const int hi  = lane >> 4;
  const float* s = src + row * NC + hi * (W / 2);
  float sum = 0.f, sum2 = 0.f;
#pragma unroll
  for (int i = 0; i < W / 2; i += 4) {
    float4 v = *(const float4*)(s + i);
    sum  += v.x + v.y + v.z + v.w;
    sum2 += v.x * v.x + v.y * v.y + v.z * v.z + v.w * v.w;
  }
  sum  += __shfl_xor(sum, 16, 32);
  sum2 += __shfl_xor(sum2, 16, 32);
  const float mean = sum * (1.0f / W);
  const float var  = sum2 * (1.0f / W) - mean * mean;
  const float inv  = rsqrtf(var + 1e-5f);

  float* d = dst + row * NC + hi * (W / 2);
  const float* gg = g    + hi * (W / 2);
  const float* bb = beta + hi * (W / 2);
#pragma unroll
  for (int i = 0; i < W / 2; i += 4) {
    float4 v = *(const float4*)(s + i);
    d[i + 0] = (v.x - mean) * inv * gg[i + 0] + bb[i + 0];
    d[i + 1] = (v.y - mean) * inv * gg[i + 1] + bb[i + 1];
    d[i + 2] = (v.z - mean) * inv * gg[i + 2] + bb[i + 2];
    d[i + 3] = (v.w - mean) * inv * gg[i + 3] + bb[i + 3];
  }
}

// ---------------------------------------------------------------------------
// 128 -> 2 logit head + cross entropy, one row per lane (lanes 0-15).
// h (LDS) and w2 (global, row-major [128][2]) are both 16B aligned -> b128.
// ---------------------------------------------------------------------------
static __device__ __forceinline__ float head_ce(
    const float* __restrict__ h, const float* __restrict__ w2,
    const float* __restrict__ b2, int label)
{
  const float4* h4 = (const float4*)h;
  const float4* w4 = (const float4*)w2;  // w4[2c] = {w[4c..4c+3]} interleaved pairs
  float l0 = b2[0], l1 = b2[1];
#pragma unroll
  for (int c = 0; c < 32; ++c) {
    const float4 hv = h4[c];
    const float4 wa = w4[2 * c + 0];   // cols 4c, 4c+1 (pairs)
    const float4 wb = w4[2 * c + 1];   // cols 4c+2, 4c+3
    l0 = fmaf(hv.x, wa.x, l0); l1 = fmaf(hv.x, wa.y, l1);
    l0 = fmaf(hv.y, wa.z, l0); l1 = fmaf(hv.y, wa.w, l1);
    l0 = fmaf(hv.z, wb.x, l0); l1 = fmaf(hv.z, wb.y, l1);
    l0 = fmaf(hv.w, wb.z, l0); l1 = fmaf(hv.w, wb.w, l1);
  }
  const float mx  = fmaxf(l0, l1);
  const float lse = mx + logf(expf(l0 - mx) + expf(l1 - mx));
  return lse - (label ? l1 : l0);
}

// ---------------------------------------------------------------------------
// Fused per-row pipeline. One wave (32 threads) per 16-row tile; all 16 rows
// belong to the same batch index b (64 rows per b, tiles are 16-aligned).
// LDS regions (cols): R0=0..127 c_rep, R1=128..255 s_rep, R2=256..383,
// R3=384..511 scratch.
// ---------------------------------------------------------------------------
__global__ __launch_bounds__(32) void fused_main(
    const float* __restrict__ xc, const float* __restrict__ xs,
    const int* __restrict__ labels,
    const u16* __restrict__ Pce1, const u16* __restrict__ Pce2,
    const u16* __restrict__ Pse1, const u16* __restrict__ Pse2,
    const u16* __restrict__ Pcd1, const u16* __restrict__ Pjd1,
    const float* __restrict__ ce_b1, const float* __restrict__ ce_b2,
    const float* __restrict__ se_b1, const float* __restrict__ se_b2,
    const float* __restrict__ cd_g,  const float* __restrict__ cd_b,
    const float* __restrict__ cd_b1,
    const float* __restrict__ cd_w2, const float* __restrict__ cd_b2,
    const float* __restrict__ jd_g,  const float* __restrict__ jd_b,
    const float* __restrict__ jd_b1,
    const float* __restrict__ jd_w2, const float* __restrict__ jd_b2,
    float* __restrict__ ceC, float* __restrict__ ceJ)
{
  __shared__ float buf[16 * NC];   // 33 KB

  const int lane = threadIdx.x;
  const int n0   = blockIdx.x * 16;
  const int b    = n0 >> 6;        // T=64 rows per batch sample
  const int row  = lane & 15;
  const int hi   = lane >> 4;
  const int label = labels[b];

  // prefetch next tile's inputs while we compute (global_prefetch_b8)
  __builtin_prefetch(xc + (size_t)(n0 + 16) * F_DIM, 0, 1);
  __builtin_prefetch(xs + (size_t)(n0 + 16) * F_DIM, 0, 1);

  // ---- causal branch ----
  {
    const float* sg = xc + (size_t)(n0 + row) * F_DIM + hi * 64;
    float* d = buf + row * NC + 256 + hi * 64;
#pragma unroll
    for (int i = 0; i < 64; i += 4) *(float4*)(d + i) = *(const float4*)(sg + i);
  }
  __syncthreads();
  wave_layer<4, true >(buf + 256, Pce1, ce_b1, buf + 384, lane); __syncthreads();
  wave_layer<4, false>(buf + 384, Pce2, ce_b2, buf + 0,   lane); __syncthreads();

  // ---- spurious branch ----
  {
    const float* sg = xs + (size_t)(n0 + row) * F_DIM + hi * 64;
    float* d = buf + row * NC + 256 + hi * 64;
#pragma unroll
    for (int i = 0; i < 64; i += 4) *(float4*)(d + i) = *(const float4*)(sg + i);
  }
  __syncthreads();
  wave_layer<4, true >(buf + 256, Pse1, se_b1, buf + 384, lane); __syncthreads();
  wave_layer<4, false>(buf + 384, Pse2, se_b2, buf + 128, lane); __syncthreads();

  // ---- causal decoder: LN(c_rep) -> GELU(linear) -> 2-logit head ----
  wave_ln<128>(buf + 0, buf + 256, cd_g, cd_b, lane); __syncthreads();
  wave_layer<4, true>(buf + 256, Pcd1, cd_b1, buf + 384, lane); __syncthreads();
  if (lane < 16) {
    ceC[n0 + lane] = head_ce(buf + lane * NC + 384, cd_w2, cd_b2, label);
  }
  __syncthreads();

  // ---- joint decoder: LN(concat) over 256 -> GELU(linear 256->128) -> head ----
  wave_ln<256>(buf + 0, buf + 256, jd_g, jd_b, lane); __syncthreads();
  wave_layer<8, true>(buf + 256, Pjd1, jd_b1, buf + 0, lane); __syncthreads();
  if (lane < 16) {
    ceJ[n0 + lane] = head_ce(buf + lane * NC + 0, jd_w2, jd_b2, label);
  }
}

// ---------------------------------------------------------------------------
// Per-timestep reductions over the batch: label counts + masked CE sums.
// ---------------------------------------------------------------------------
__global__ __launch_bounds__(256) void reduce_t(
    const float* __restrict__ ceC, const float* __restrict__ ceJ,
    const int* __restrict__ mask, const int* __restrict__ labels,
    float* __restrict__ counts, float* __restrict__ Sc, float* __restrict__ Sj)
{
  const int t   = blockIdx.x;
  const int tid = threadIdx.x;
  float c0 = 0.f, c1 = 0.f, sc = 0.f, sj = 0.f;
  for (int b = tid; b < B_DIM; b += 256) {
    const float m = mask[b * T_DIM + t] ? 1.0f : 0.0f;
    if (labels[b] == 0) c0 += m; else c1 += m;
    sc += m * ceC[b * T_DIM + t];
    sj += m * ceJ[b * T_DIM + t];
  }
  __shared__ float s0[256], s1[256], s2[256], s3[256];
  s0[tid] = c0; s1[tid] = c1; s2[tid] = sc; s3[tid] = sj;
  __syncthreads();
  for (int off = 128; off > 0; off >>= 1) {
    if (tid < off) {
      s0[tid] += s0[tid + off]; s1[tid] += s1[tid + off];
      s2[tid] += s2[tid + off]; s3[tid] += s3[tid + off];
    }
    __syncthreads();
  }
  if (tid == 0) {
    counts[2 * t + 0] = s0[0];
    counts[2 * t + 1] = s1[0];
    Sc[t] = s2[0];
    Sj[t] = s3[0];
  }
}

// ---------------------------------------------------------------------------
// Per-step entropy / validity + aux_loss scalar (single tiny block).
// ---------------------------------------------------------------------------
__global__ __launch_bounds__(64) void finalize_t(
    const float* __restrict__ counts, const float* __restrict__ Sc,
    const float* __restrict__ Sj,
    float* __restrict__ ent, float* __restrict__ sv, float* __restrict__ aux_out)
{
  const int t = threadIdx.x;
  __shared__ float sA[64], sN[64];
  const float c0 = counts[2 * t], c1 = counts[2 * t + 1];
  const float tot   = c0 + c1;
  const float denom = fmaxf(tot, 1.0f);
  const float p0 = c0 / denom, p1 = c1 / denom;
  const float e  = -(p0 * logf(p0 + 1e-8f) + p1 * logf(p1 + 1e-8f));
  const float v  = (tot >= 2.0f) ? 1.0f : 0.0f;
  ent[t] = e; sv[t] = v;
  const float aux_step = 0.5f * (Sc[t] + Sj[t]) / denom;
  sA[t] = v * aux_step; sN[t] = v;
  __syncthreads();
  for (int off = 32; off > 0; off >>= 1) {
    if (t < off) { sA[t] += sA[t + off]; sN[t] += sN[t + off]; }
    __syncthreads();
  }
  if (t == 0) {
    const float nv = sN[0];
    aux_out[0] = (nv > 0.0f) ? (sA[0] / fmaxf(nv, 1.0f)) : 0.0f;
  }
}

// ---------------------------------------------------------------------------
// Elementwise reward over [B,T].
// ---------------------------------------------------------------------------
__global__ __launch_bounds__(256) void reward_k(
    const float* __restrict__ ceC, const float* __restrict__ ceJ,
    const int* __restrict__ mask,
    const float* __restrict__ ent, const float* __restrict__ sv,
    float* __restrict__ out)
{
  const int idx = blockIdx.x * 256 + threadIdx.x;
  if (idx >= NROW) return;
  const int t = idx & (T_DIM - 1);
  const float m    = mask[idx] ? 1.0f : 0.0f;
  const float cc   = ceC[idx], cj = ceJ[idx];
  const float mi   = ent[t] - cc;
  const float cond = fmaxf(cc - cj, 0.0f);
  out[idx] = m * sv[t] * (mi - LAM * cond);
}

// ---------------------------------------------------------------------------
// Launch: repack weights once per call, run fused pipeline, then reductions.
// Workspace use ~2.3 MB.
// ---------------------------------------------------------------------------
extern "C" void kernel_launch(void* const* d_in, const int* in_sizes, int n_in,
                              void* d_out, int out_size, void* d_ws, size_t ws_size,
                              hipStream_t stream)
{
  (void)in_sizes; (void)n_in; (void)out_size; (void)ws_size;
  const float* xc     = (const float*)d_in[0];
  const float* xs     = (const float*)d_in[1];
  const int*   labels = (const int*)d_in[2];
  const int*   mask   = (const int*)d_in[3];
  const float* ce_w1 = (const float*)d_in[4];  const float* ce_b1 = (const float*)d_in[5];
  const float* ce_w2 = (const float*)d_in[6];  const float* ce_b2 = (const float*)d_in[7];
  const float* se_w1 = (const float*)d_in[8];  const float* se_b1 = (const float*)d_in[9];
  const float* se_w2 = (const float*)d_in[10]; const float* se_b2 = (const float*)d_in[11];
  const float* cd_g  = (const float*)d_in[12]; const float* cd_b  = (const float*)d_in[13];
  const float* cd_w1 = (const float*)d_in[14]; const float* cd_b1 = (const float*)d_in[15];
  const float* cd_w2 = (const float*)d_in[16]; const float* cd_b2 = (const float*)d_in[17];
  const float* jd_g  = (const float*)d_in[18]; const float* jd_b  = (const float*)d_in[19];
  const float* jd_w1 = (const float*)d_in[20]; const float* jd_b1 = (const float*)d_in[21];
  const float* jd_w2 = (const float*)d_in[22]; const float* jd_b2 = (const float*)d_in[23];

  char* ws = (char*)d_ws;
  size_t off = 0;
  u16* Pce1 = (u16*)(ws + off); off += 128 * 128 * 2;
  u16* Pce2 = (u16*)(ws + off); off += 128 * 128 * 2;
  u16* Pse1 = (u16*)(ws + off); off += 128 * 128 * 2;
  u16* Pse2 = (u16*)(ws + off); off += 128 * 128 * 2;
  u16* Pcd1 = (u16*)(ws + off); off += 128 * 128 * 2;
  u16* Pjd1 = (u16*)(ws + off); off += 256 * 128 * 2;
  float* ceC    = (float*)(ws + off); off += (size_t)NROW * 4;
  float* ceJ    = (float*)(ws + off); off += (size_t)NROW * 4;
  float* counts = (float*)(ws + off); off += 128 * 4;
  float* Sc     = (float*)(ws + off); off += 64 * 4;
  float* Sj     = (float*)(ws + off); off += 64 * 4;
  float* ent    = (float*)(ws + off); off += 64 * 4;
  float* sv     = (float*)(ws + off); off += 64 * 4;

  pack_weight<<<64,  256, 0, stream>>>(ce_w1, Pce1, 4);
  pack_weight<<<64,  256, 0, stream>>>(ce_w2, Pce2, 4);
  pack_weight<<<64,  256, 0, stream>>>(se_w1, Pse1, 4);
  pack_weight<<<64,  256, 0, stream>>>(se_w2, Pse2, 4);
  pack_weight<<<64,  256, 0, stream>>>(cd_w1, Pcd1, 4);
  pack_weight<<<128, 256, 0, stream>>>(jd_w1, Pjd1, 8);

  fused_main<<<NROW / 16, 32, 0, stream>>>(
      xc, xs, labels,
      Pce1, Pce2, Pse1, Pse2, Pcd1, Pjd1,
      ce_b1, ce_b2, se_b1, se_b2,
      cd_g, cd_b, cd_b1, cd_w2, cd_b2,
      jd_g, jd_b, jd_b1, jd_w2, jd_b2,
      ceC, ceJ);

  reduce_t<<<T_DIM, 256, 0, stream>>>(ceC, ceJ, mask, labels, counts, Sc, Sj);
  finalize_t<<<1, 64, 0, stream>>>(counts, Sc, Sj, ent, sv, ((float*)d_out) + NROW);
  reward_k<<<NROW / 256, 256, 0, stream>>>(ceC, ceJ, mask, ent, sv, (float*)d_out);
}
